// VanillaAttention_68410239090699
// MI455X (gfx1250) — matrix operand order repaired and checked by reading, not compiled, not run on hardware
//
#include <hip/hip_runtime.h>
#include <hip/hip_bf16.h>
#include <math.h>

// Problem constants (B, S, D) from the reference
#define B_ 4
#define S_ 4096
#define D_ 1024

typedef __bf16 bf16_t;
typedef __attribute__((ext_vector_type(16))) __bf16 v16bf;
typedef __attribute__((ext_vector_type(8)))  __bf16 v8bf;
typedef __attribute__((ext_vector_type(8)))  float  v8f;

union V16U { v16bf v; v8bf h[2]; };

__device__ __forceinline__ v8f wmma_bf16(v16bf a, v16bf b, v8f c) {
  // D = A(16x32 bf16) * B(32x16 bf16) + C(16x16 f32)
  return __builtin_amdgcn_wmma_f32_16x16x32_bf16(false, a, false, b,
                                                 (short)0, c, false, false);
}

// Async global->LDS copy, 16B per lane (tracked by ASYNCcnt).
__device__ __forceinline__ void async_copy_b128(unsigned lds_off, const void* gptr) {
  asm volatile("global_load_async_to_lds_b128 %0, %1, off"
               :: "v"(lds_off), "v"(gptr) : "memory");
}
__device__ __forceinline__ void wait_async0() {
  asm volatile("s_wait_asynccnt 0" ::: "memory");
}
// Low 32 bits of a generic pointer to __shared__ == LDS byte address.
__device__ __forceinline__ unsigned lds_addr(const void* p) {
  return (unsigned)(unsigned long long)p;
}

// ---------------------------------------------------------------------------
// Kernel 0: bulk f32 -> bf16 conversion (one pass; keeps GEMM loops cvt-free)
// ---------------------------------------------------------------------------
__global__ __launch_bounds__(256) void cvt_bf16_kernel(
    const float* __restrict__ src, bf16_t* __restrict__ dst, int n8) {
  const int i = blockIdx.x * blockDim.x + threadIdx.x;
  if (i < n8) {
    v8f f = *(const v8f*)(src + (size_t)i * 8);
    v8bf o;
    #pragma unroll
    for (int j = 0; j < 8; ++j) o[j] = (__bf16)f[j];
    *(v8bf*)(dst + (size_t)i * 8) = o;
  }
}

// ---------------------------------------------------------------------------
// Kernel 1: Y = x @ W^T (bf16 WMMA), fused RoPE for Q/K.
//   which = blockIdx.z: 0 -> Q (rope), 1 -> K (rope), 2 -> V (stored transposed)
// Block = 4 waves; x-tile (16x1024 bf16, 32KB) staged once in LDS via async
// copy; each wave computes a 16x64 strip. B-frags preloaded into distinct
// registers so 4 loads stay in flight per chunk.
// ---------------------------------------------------------------------------
__global__ __launch_bounds__(128) void proj_rope_kernel(
    const bf16_t* __restrict__ xb, const bf16_t* __restrict__ Wb,
    bf16_t* __restrict__ Qb, bf16_t* __restrict__ Kb, bf16_t* __restrict__ Vt)
{
  __shared__ __align__(16) bf16_t Xs[16 * D_];   // 32 KB

  const int lane = threadIdx.x & 31;
  const int wave = threadIdx.x >> 5;
  const int hs   = lane >> 4;
  const int l15  = lane & 15;
  const int row0 = blockIdx.x * 16;
  const int col0 = blockIdx.y * 256 + wave * 64;
  const int which = blockIdx.z;
  const bf16_t* W = Wb + (size_t)which * D_ * D_;

  // Stage x tile (rows row0..row0+15 are one contiguous 32KB block)
  {
    const bf16_t* xsrc = xb + (size_t)row0 * D_;
    #pragma unroll
    for (int i = 0; i < 16; ++i) {
      const int e8 = (i * 128 + (int)threadIdx.x) * 8;
      async_copy_b128(lds_addr(Xs + e8), xsrc + e8);
    }
    wait_async0();
  }
  __syncthreads();

  v8f acc[4];
  #pragma unroll
  for (int n = 0; n < 4; ++n) acc[n] = (v8f){};

  #pragma unroll 4
  for (int c = 0; c < D_ / 32; ++c) {
    const int dlo = 32 * c + 16 * hs;
    V16U a;
    a.h[0] = *(const v8bf*)(Xs + l15 * D_ + dlo);
    a.h[1] = *(const v8bf*)(Xs + l15 * D_ + dlo + 8);
    const int db = 32 * c + 16 * hs;
    v16bf bb[4];
    #pragma unroll
    for (int n = 0; n < 4; ++n) {
      const int wrow = col0 + 16 * n + l15;      // B column == W row
      bb[n] = *(const v16bf*)(W + (size_t)wrow * D_ + db);
    }
    #pragma unroll
    for (int n = 0; n < 4; ++n) acc[n] = wmma_bf16(a.v, bb[n], acc[n]);
  }

  // Epilogue: RoPE (Q,K) + store. C layout: elem (M = v + 8*hs, N = col+l15).
  #pragma unroll
  for (int n = 0; n < 4; ++n) {
    const int ncol  = col0 + 16 * n + l15;
    const int pairI = ncol >> 1;
    const bool even = (ncol & 1) == 0;
    const float invfreq =
        __expf(-(float)(2 * pairI) * (9.210340371976184f / (float)D_));
    #pragma unroll
    for (int v = 0; v < 8; ++v) {
      const int r = row0 + v + 8 * hs;           // global row = b*S + s
      float val = acc[n][v];
      if (which < 2) {
        const float partner = __shfl_xor(val, 1, 32);
        const int   s   = r & (S_ - 1);
        const float ang = (float)s * invfreq;
        float sn, cs;
        __sincosf(ang, &sn, &cs);
        val = even ? (val * cs - partner * sn) : (partner * sn + val * cs);
      }
      if (which == 0) {
        Qb[(size_t)r * D_ + ncol] = (bf16_t)val;
      } else if (which == 1) {
        Kb[(size_t)r * D_ + ncol] = (bf16_t)val;
      } else {
        const int bidx = r >> 12;                // r / S_
        const int s    = r & (S_ - 1);
        Vt[((size_t)bidx * D_ + ncol) * S_ + s] = (bf16_t)val;
      }
    }
  }
}

// ---------------------------------------------------------------------------
// Kernel 2: flash attention, causal, bf16 WMMA + online softmax (f32 state).
// 8 waves / block, 32 query rows / block (two 16-row A-subtiles per wave so
// every K/V B-fragment feeds two WMMAs). Wave w owns:
//   - score N-subtile w (2 x 16q x 16k of the 128-wide K tile)
//   - output d-slice [128w, 128w+128): 2 x (16x128) f32 accumulators
// ---------------------------------------------------------------------------
__global__ __launch_bounds__(256) void flash_attn_kernel(
    const bf16_t* __restrict__ Qb, const bf16_t* __restrict__ Kb,
    const bf16_t* __restrict__ Vt, float* __restrict__ out)
{
  __shared__ __align__(16) bf16_t Qs[32 * D_];      // 64 KB: Q tile
  __shared__ __align__(16) bf16_t Pbuf[32 * 128];   //  8 KB: P tile (bf16)
  __shared__ float redmax[2][8][16];
  __shared__ float redsum[2][8][16];

  const int tid  = threadIdx.x;
  const int lane = tid & 31;
  const int wave = tid >> 5;
  const int hs   = lane >> 4;
  const int l15  = lane & 15;
  const int b    = blockIdx.y;
  const int q0   = blockIdx.x * 32;

  // Stage Q tile via async copy (contiguous 64KB block)
  {
    const bf16_t* qsrc = Qb + ((size_t)b * S_ + q0) * D_;
    #pragma unroll
    for (int i = 0; i < 16; ++i) {
      const int e8 = (i * 256 + tid) * 8;
      async_copy_b128(lds_addr(Qs + e8), qsrc + e8);
    }
    wait_async0();
  }
  __syncthreads();

  v8f acc[2][8];
  #pragma unroll
  for (int t = 0; t < 2; ++t)
    #pragma unroll
    for (int n = 0; n < 8; ++n) acc[t][n] = (v8f){};
  float m_run[2][8], l_run[2][8];
  #pragma unroll
  for (int t = 0; t < 2; ++t)
    #pragma unroll
    for (int v = 0; v < 8; ++v) { m_run[t][v] = -1e30f; l_run[t][v] = 0.0f; }

  const int nIter = (q0 + 32 + 127) >> 7;   // K tiles of 128, causal bound
  for (int it = 0; it < nIter; ++it) {
    const int k0 = it * 128;
    const int kw = k0 + wave * 16;          // this wave's 16 K rows

    // ---- scores: per wave 2 x (16q x 16k); one K B-frag -> 2 WMMAs ----
    v8f sc[2];
    sc[0] = (v8f){}; sc[1] = (v8f){};
    const bf16_t* krow = Kb + ((size_t)b * S_ + (kw + l15)) * D_;
    const bool pf = (it + 1) < nIter;
    #pragma unroll 4
    for (int c = 0; c < 32; ++c) {
      const int dlo = 32 * c + 16 * hs;
      V16U a0, a1;
      a0.h[0] = *(const v8bf*)(Qs + l15 * D_ + dlo);
      a0.h[1] = *(const v8bf*)(Qs + l15 * D_ + dlo + 8);
      a1.h[0] = *(const v8bf*)(Qs + (16 + l15) * D_ + dlo);
      a1.h[1] = *(const v8bf*)(Qs + (16 + l15) * D_ + dlo + 8);
      const v16bf bb = *(const v16bf*)(krow + 32 * c + 16 * hs);
      if (pf && !(c & 1))   // prefetch next K tile, 128B strides
        __builtin_prefetch(krow + (size_t)128 * D_ + 32 * c, 0, 1);
      sc[0] = wmma_bf16(a0.v, bb, sc[0]);
      sc[1] = wmma_bf16(a1.v, bb, sc[1]);
    }

    // ---- scale, causal mask, per-row max (shfl within 16-lane group) ----
    const int kcol = kw + l15;
    #pragma unroll
    for (int t = 0; t < 2; ++t) {
      #pragma unroll
      for (int v = 0; v < 8; ++v) {
        float s = sc[t][v] * 0.03125f;            // 1/sqrt(1024)
        const int qr = q0 + 16 * t + v + 8 * hs;
        if (kcol > qr) s = -1e30f;
        sc[t][v] = s;
        float m = s;
        m = fmaxf(m, __shfl_xor(m, 1, 32));
        m = fmaxf(m, __shfl_xor(m, 2, 32));
        m = fmaxf(m, __shfl_xor(m, 4, 32));
        m = fmaxf(m, __shfl_xor(m, 8, 32));
        if (l15 == v) redmax[t][wave][v + 8 * hs] = m;
      }
    }
    __syncthreads();

    // ---- combine maxes, exponentiate, row sums, write P, rescale acc ----
    #pragma unroll
    for (int t = 0; t < 2; ++t) {
      #pragma unroll
      for (int v = 0; v < 8; ++v) {
        const int r = v + 8 * hs;
        float cm = m_run[t][v];
        #pragma unroll
        for (int w2 = 0; w2 < 8; ++w2) cm = fmaxf(cm, redmax[t][w2][r]);
        const float p = __expf(sc[t][v] - cm);
        Pbuf[(16 * t + r) * 128 + wave * 16 + l15] = (bf16_t)p;
        float rs = p;
        rs += __shfl_xor(rs, 1, 32);
        rs += __shfl_xor(rs, 2, 32);
        rs += __shfl_xor(rs, 4, 32);
        rs += __shfl_xor(rs, 8, 32);
        if (l15 == v) redsum[t][wave][r] = rs;
        const float alpha = __expf(m_run[t][v] - cm);
        m_run[t][v] = cm;
        l_run[t][v] *= alpha;
        #pragma unroll
        for (int n = 0; n < 8; ++n) acc[t][n][v] *= alpha;
      }
    }
    __syncthreads();

    #pragma unroll
    for (int t = 0; t < 2; ++t) {
      #pragma unroll
      for (int v = 0; v < 8; ++v) {
        const int r = v + 8 * hs;
        float rs = 0.0f;
        #pragma unroll
        for (int w2 = 0; w2 < 8; ++w2) rs += redsum[t][w2][r];
        l_run[t][v] += rs;
      }
    }

    // ---- PV: acc[t](16x128) += P[t](16x128) * V(128 x 128-slice) ----
    #pragma unroll
    for (int c = 0; c < 4; ++c) {
      const int dlo = 32 * c + 16 * hs;
      V16U a0, a1;
      a0.h[0] = *(const v8bf*)(Pbuf + l15 * 128 + dlo);
      a0.h[1] = *(const v8bf*)(Pbuf + l15 * 128 + dlo + 8);
      a1.h[0] = *(const v8bf*)(Pbuf + (16 + l15) * 128 + dlo);
      a1.h[1] = *(const v8bf*)(Pbuf + (16 + l15) * 128 + dlo + 8);
      const int kidx = k0 + 32 * c + 16 * hs;
      #pragma unroll
      for (int n = 0; n < 8; ++n) {
        const int dcol = 128 * wave + 16 * n + l15;
        const v16bf bb =
            *(const v16bf*)(Vt + ((size_t)b * D_ + dcol) * S_ + kidx);
        acc[0][n] = wmma_bf16(a0.v, bb, acc[0][n]);
        acc[1][n] = wmma_bf16(a1.v, bb, acc[1][n]);
      }
    }
  }

  // ---- normalize + store (f32 output) ----
  #pragma unroll
  for (int t = 0; t < 2; ++t) {
    #pragma unroll
    for (int v = 0; v < 8; ++v) {
      const float inv = 1.0f / l_run[t][v];
      const int qr = q0 + 16 * t + v + 8 * hs;
      float* op = out + ((size_t)b * S_ + qr) * D_;
      #pragma unroll
      for (int n = 0; n < 8; ++n)
        op[128 * wave + 16 * n + l15] = acc[t][n][v] * inv;
    }
  }
}

// ---------------------------------------------------------------------------
extern "C" void kernel_launch(void* const* d_in, const int* in_sizes, int n_in,
                              void* d_out, int out_size, void* d_ws, size_t ws_size,
                              hipStream_t stream) {
  const float* x  = (const float*)d_in[0];
  const float* Wq = (const float*)d_in[1];
  const float* Wk = (const float*)d_in[2];
  const float* Wv = (const float*)d_in[3];
  float* out = (float*)d_out;

  const size_t elems = (size_t)B_ * S_ * D_;   // 16,777,216
  const size_t wel   = (size_t)D_ * D_;        //  1,048,576
  bf16_t* Qb = (bf16_t*)d_ws;                  // [B][S][D] bf16, rope'd
  bf16_t* Kb = Qb + elems;                     // [B][S][D] bf16, rope'd
  bf16_t* Vt = Kb + elems;                     // [B][D][S] bf16, transposed
  bf16_t* xb = Vt + elems;                     // [B][S][D] bf16
  bf16_t* Wb = xb + elems;                     // [3][D][D] bf16

  // f32 -> bf16 once
  cvt_bf16_kernel<<<(int)(elems / 8 / 256), 256, 0, stream>>>(x, xb, (int)(elems / 8));
  cvt_bf16_kernel<<<(int)(wel / 8 / 256), 256, 0, stream>>>(Wq, Wb, (int)(wel / 8));
  cvt_bf16_kernel<<<(int)(wel / 8 / 256), 256, 0, stream>>>(Wk, Wb + wel, (int)(wel / 8));
  cvt_bf16_kernel<<<(int)(wel / 8 / 256), 256, 0, stream>>>(Wv, Wb + 2 * wel, (int)(wel / 8));

  // Projections + RoPE
  dim3 gProj((B_ * S_) / 16, D_ / 256, 3);
  proj_rope_kernel<<<gProj, 128, 0, stream>>>(xb, Wb, Qb, Kb, Vt);

  // Flash attention: one block per (32 query rows, batch)
  dim3 gAttn(S_ / 32, B_, 1);
  flash_attn_kernel<<<gAttn, 256, 0, stream>>>(Qb, Kb, Vt, out);
}
